// GraphCastProcessor_77068893159639
// MI455X (gfx1250) — compile-verified, matrix operand order checked by reading, not run on hardware
//
#include <hip/hip_runtime.h>

#define DF 64
#define LAYERS 4
#define LN_EPS 1e-5f

typedef __attribute__((ext_vector_type(16))) __bf16 v16bf;
typedef __attribute__((ext_vector_type(8)))  float  v8f;

union FragAB { unsigned u[8]; uint4 q[2]; v16bf v; };

__device__ __forceinline__ unsigned short f2bf(float f) {
    unsigned u = __float_as_uint(f);
    u += 0x7FFFu + ((u >> 16) & 1u);          // round-to-nearest-even
    return (unsigned short)(u >> 16);
}
__device__ __forceinline__ unsigned pack2(float lo, float hi) {
    return (unsigned)f2bf(lo) | ((unsigned)f2bf(hi) << 16);
}

// ---------------------------------------------------------------------------
// Shared 16xKIN @ KINx64 -> silu -> 64x64 -> +b2 -> LayerNorm MLP tile.
// A tile in LDS as bf16 bits (row stride LDK ushorts); pairs of consecutive K
// are contiguous, so each half-fragment is one aligned ds_load_b128.
// Weights in LDS in fragment-major layout: frag (t,nt) at [(t*4+nt)*256],
// lane L's 8 uints contiguous at [.. + L*8]  -> two ds_load_b128 per B frag.
// Output: out[nt][r] for element (m = r + 8*(lane>>4), n = nt*16 + (lane&15)).
// ---------------------------------------------------------------------------
template <int KIN, int LDK>
__device__ __forceinline__ void mlp_tile(
    const unsigned short* As, int row0, unsigned short* Hs,
    const unsigned* W1p, const unsigned* W2p,
    const float* b1, const float* b2, const float* gg, const float* bb,
    float (&out)[4][8])
{
    const int lane = threadIdx.x & 31;
    const int n15  = lane & 15;
    const int hh   = lane >> 4;
    const int m    = n15;

    v8f zero;
#pragma unroll
    for (int j = 0; j < 8; ++j) zero[j] = 0.0f;

    v8f acc[4];
#pragma unroll
    for (int nt = 0; nt < 4; ++nt) acc[nt] = zero;

    // ---- GEMM1: [16,KIN] @ [KIN,64] ----
#pragma unroll
    for (int t = 0; t < KIN / 32; ++t) {
        FragAB a;
        a.q[0] = *(const uint4*)&As[(row0 + m) * LDK + t * 32 + hh * 8];
        a.q[1] = *(const uint4*)&As[(row0 + m) * LDK + t * 32 + 16 + hh * 8];
#pragma unroll
        for (int nt = 0; nt < 4; ++nt) {
            FragAB b;
            const uint4* bp = (const uint4*)&W1p[(t * 4 + nt) * 256 + lane * 8];
            b.q[0] = bp[0];
            b.q[1] = bp[1];
            acc[nt] = __builtin_amdgcn_wmma_f32_16x16x32_bf16(
                false, a.v, false, b.v, (short)0, acc[nt], false, false);
        }
    }

    // ---- bias + SiLU, restage h (reuses consumed A columns 0..63) ----
#pragma unroll
    for (int nt = 0; nt < 4; ++nt) {
        const int n = nt * 16 + n15;
        const float bias = b1[n];
#pragma unroll
        for (int r = 0; r < 8; ++r) {
            const int mm = r + 8 * hh;
            const float x = acc[nt][r] + bias;
            const float s = x / (1.0f + __expf(-x));
            Hs[mm * LDK + n] = f2bf(s);
        }
    }
    __syncthreads();

    // ---- GEMM2: [16,64] @ [64,64] ----
    v8f acc2[4];
#pragma unroll
    for (int nt = 0; nt < 4; ++nt) acc2[nt] = zero;
#pragma unroll
    for (int t = 0; t < 2; ++t) {
        FragAB a;
        a.q[0] = *(const uint4*)&Hs[m * LDK + t * 32 + hh * 8];
        a.q[1] = *(const uint4*)&Hs[m * LDK + t * 32 + 16 + hh * 8];
#pragma unroll
        for (int nt = 0; nt < 4; ++nt) {
            FragAB b;
            const uint4* bp = (const uint4*)&W2p[(t * 4 + nt) * 256 + lane * 8];
            b.q[0] = bp[0];
            b.q[1] = bp[1];
            acc2[nt] = __builtin_amdgcn_wmma_f32_16x16x32_bf16(
                false, a.v, false, b.v, (short)0, acc2[nt], false, false);
        }
    }

    // ---- +b2, LayerNorm across 64 columns (lanes within 16-lane half) ----
    float vals[4][8];
#pragma unroll
    for (int nt = 0; nt < 4; ++nt) {
        const float b2n = b2[nt * 16 + n15];
#pragma unroll
        for (int r = 0; r < 8; ++r) vals[nt][r] = acc2[nt][r] + b2n;
    }
#pragma unroll
    for (int r = 0; r < 8; ++r) {
        float s = vals[0][r] + vals[1][r] + vals[2][r] + vals[3][r];
        float q = vals[0][r] * vals[0][r] + vals[1][r] * vals[1][r] +
                  vals[2][r] * vals[2][r] + vals[3][r] * vals[3][r];
#pragma unroll
        for (int msk = 1; msk < 16; msk <<= 1) {
            s += __shfl_xor(s, msk, 16);
            q += __shfl_xor(q, msk, 16);
        }
        const float mean = s * (1.0f / 64.0f);
        float var = q * (1.0f / 64.0f) - mean * mean;
        const float rstd = rsqrtf(var + LN_EPS);
#pragma unroll
        for (int nt = 0; nt < 4; ++nt) {
            const int n = nt * 16 + n15;
            out[nt][r] = (vals[nt][r] - mean) * rstd * gg[n] + bb[n];
        }
    }
}

// ---------------------------------------------------------------------------
// Edge kernel: 64 edges / block (128 threads, 4 waves, 16 edges per wave).
// ---------------------------------------------------------------------------
__global__ void __launch_bounds__(128)
edge_kernel(const long long* __restrict__ ei, const unsigned short* __restrict__ x_bf,
            float* __restrict__ e_out, float* __restrict__ agg,
            const unsigned* __restrict__ w1p_g, const unsigned* __restrict__ w2p_g,
            const float* __restrict__ b1g, const float* __restrict__ b2g,
            const float* __restrict__ gg_g, const float* __restrict__ beg,
            long long Ecount)
{
    __shared__ __align__(16) unsigned short As[64 * 192];
    __shared__ __align__(16) unsigned W1s[24 * 256];   // 6 ktiles x 4 ntiles
    __shared__ __align__(16) unsigned W2s[8 * 256];    // 2 ktiles x 4 ntiles
    __shared__ float b1s[64], b2s[64], gs[64], bes[64];
    __shared__ int sends[64], recvs[64];

    const int tid = threadIdx.x;
    const long long ebase = (long long)blockIdx.x * 64;

    if (tid < 64) {
        long long e = ebase + tid;
        if (e >= Ecount) e = Ecount - 1;
        sends[tid] = (int)ei[e];
        recvs[tid] = (int)ei[Ecount + e];
        b1s[tid] = b1g[tid]; b2s[tid] = b2g[tid];
        gs[tid]  = gg_g[tid]; bes[tid] = beg[tid];
    }
    {   // stage frag-major weights (6144 + 2048 uints) as b128 copies
        const uint4* s1 = (const uint4*)w1p_g; uint4* d1 = (uint4*)W1s;
#pragma unroll
        for (int i = 0; i < 12; ++i) d1[tid + 128 * i] = s1[tid + 128 * i];
        const uint4* s2 = (const uint4*)w2p_g; uint4* d2 = (uint4*)W2s;
#pragma unroll
        for (int i = 0; i < 4; ++i) d2[tid + 128 * i] = s2[tid + 128 * i];
    }
    __syncthreads();

    {   // stage A tile: [64 edges][x_send(64) | x_recv(64) | edge_attr(64)] bf16
        const int row = tid >> 1, half = tid & 1;
        long long e = ebase + row;
        if (e >= Ecount) e = Ecount - 1;
        const uint4* xs = (const uint4*)(x_bf + (long long)sends[row] * DF + half * 32);
        uint4* dA = (uint4*)&As[row * 192 + half * 32];
        dA[0] = xs[0]; dA[1] = xs[1]; dA[2] = xs[2]; dA[3] = xs[3];
        const uint4* xr = (const uint4*)(x_bf + (long long)recvs[row] * DF + half * 32);
        uint4* dB = (uint4*)&As[row * 192 + 64 + half * 32];
        dB[0] = xr[0]; dB[1] = xr[1]; dB[2] = xr[2]; dB[3] = xr[3];
        const float4* ef = (const float4*)(e_out + e * DF + half * 32);
        unsigned* dE = (unsigned*)&As[row * 192 + 128 + half * 32];
#pragma unroll
        for (int j = 0; j < 8; ++j) {
            const float4 f = ef[j];
            dE[2 * j]     = pack2(f.x, f.y);
            dE[2 * j + 1] = pack2(f.z, f.w);
        }
    }
    __syncthreads();

    const int wave = tid >> 5;
    const int row0 = wave * 16;
    float out[4][8];
    mlp_tile<192, 192>(As, row0, As + row0 * 192, W1s, W2s, b1s, b2s, gs, bes, out);

    // stage LN output to LDS as fp32 rows (A tile is dead now); 96 floats/row
    float* Fs = (float*)As;
    const int lane = tid & 31, n15 = lane & 15, hh = lane >> 4;
#pragma unroll
    for (int r = 0; r < 8; ++r) {
        const int mm = r + 8 * hh;
#pragma unroll
        for (int nt = 0; nt < 4; ++nt)
            Fs[(row0 + mm) * 96 + nt * 16 + n15] = out[nt][r];
    }
    __syncthreads();

    // vectorized epilogue: edge residual RMW (b128) + scatter-add into agg
    {
        const int rrow = lane >> 1, half = lane & 1;
        const long long e = ebase + row0 + rrow;
        if (e < Ecount) {
            const long long rv = recvs[row0 + rrow];
            const float4* fs4 = (const float4*)&Fs[(row0 + rrow) * 96 + half * 32];
            float4* pe4 = (float4*)(e_out + e * DF + half * 32);
            float* pa = agg + rv * DF + half * 32;
#pragma unroll
            for (int j = 0; j < 8; ++j) {
                const float4 s = fs4[j];
                float4 d = pe4[j];
                d.x += s.x; d.y += s.y; d.z += s.z; d.w += s.w;
                pe4[j] = d;
                atomicAdd(pa + 4 * j + 0, s.x);
                atomicAdd(pa + 4 * j + 1, s.y);
                atomicAdd(pa + 4 * j + 2, s.z);
                atomicAdd(pa + 4 * j + 3, s.w);
            }
        }
    }
}

// ---------------------------------------------------------------------------
// Node kernel: 64 nodes / block. Also refreshes bf16 mirror + zeroes agg.
// ---------------------------------------------------------------------------
__global__ void __launch_bounds__(128)
node_kernel(float* __restrict__ x_out, unsigned short* __restrict__ x_bf,
            float* __restrict__ agg,
            const unsigned* __restrict__ w1p_g, const unsigned* __restrict__ w2p_g,
            const float* __restrict__ b1g, const float* __restrict__ b2g,
            const float* __restrict__ gg_g, const float* __restrict__ beg,
            long long Ncount)
{
    __shared__ __align__(16) unsigned short As[64 * 128];
    __shared__ __align__(16) unsigned W1s[16 * 256];   // 4 ktiles x 4 ntiles
    __shared__ __align__(16) unsigned W2s[8 * 256];
    __shared__ float b1s[64], b2s[64], gs[64], bes[64];

    const int tid = threadIdx.x;
    const long long nbase = (long long)blockIdx.x * 64;

    if (tid < 64) {
        b1s[tid] = b1g[tid]; b2s[tid] = b2g[tid];
        gs[tid]  = gg_g[tid]; bes[tid] = beg[tid];
    }
    {
        const uint4* s1 = (const uint4*)w1p_g; uint4* d1 = (uint4*)W1s;
#pragma unroll
        for (int i = 0; i < 8; ++i) d1[tid + 128 * i] = s1[tid + 128 * i];
        const uint4* s2 = (const uint4*)w2p_g; uint4* d2 = (uint4*)W2s;
#pragma unroll
        for (int i = 0; i < 4; ++i) d2[tid + 128 * i] = s2[tid + 128 * i];
    }
    {   // stage A tile: [64 nodes][x(64) | agg(64)] bf16
        const int row = tid >> 1, half = tid & 1;
        long long node = nbase + row;
        if (node >= Ncount) node = Ncount - 1;
        const float4* xf = (const float4*)(x_out + node * DF + half * 32);
        unsigned* d0 = (unsigned*)&As[row * 128 + half * 32];
#pragma unroll
        for (int j = 0; j < 8; ++j) {
            const float4 f = xf[j];
            d0[2 * j] = pack2(f.x, f.y); d0[2 * j + 1] = pack2(f.z, f.w);
        }
        const float4* af = (const float4*)(agg + node * DF + half * 32);
        unsigned* d1 = (unsigned*)&As[row * 128 + 64 + half * 32];
#pragma unroll
        for (int j = 0; j < 8; ++j) {
            const float4 f = af[j];
            d1[2 * j] = pack2(f.x, f.y); d1[2 * j + 1] = pack2(f.z, f.w);
        }
    }
    __syncthreads();

    const int wave = tid >> 5;
    const int row0 = wave * 16;
    float out[4][8];
    mlp_tile<128, 128>(As, row0, As + row0 * 128, W1s, W2s, b1s, b2s, gs, bes, out);

    // stage LN output to LDS as fp32 rows (64 floats/row, full A-tile row)
    float* Fs = (float*)As;
    const int lane = tid & 31, n15 = lane & 15, hh = lane >> 4;
#pragma unroll
    for (int r = 0; r < 8; ++r) {
        const int mm = r + 8 * hh;
#pragma unroll
        for (int nt = 0; nt < 4; ++nt)
            Fs[(row0 + mm) * 64 + nt * 16 + n15] = out[nt][r];
    }
    __syncthreads();

    // vectorized epilogue: x residual RMW, bf16 mirror refresh, agg clear
    {
        const int rrow = lane >> 1, half = lane & 1;
        const long long node = nbase + row0 + rrow;
        if (node < Ncount) {
            const float4* fs4 = (const float4*)&Fs[(row0 + rrow) * 64 + half * 32];
            float4* px = (float4*)(x_out + node * DF + half * 32);
            float4* pa = (float4*)(agg + node * DF + half * 32);
            uint4* pb = (uint4*)(x_bf + node * DF + half * 32);
            const float4 z4 = make_float4(0.0f, 0.0f, 0.0f, 0.0f);
            unsigned ub[16];
#pragma unroll
            for (int j = 0; j < 8; ++j) {
                const float4 s = fs4[j];
                float4 d = px[j];
                d.x += s.x; d.y += s.y; d.z += s.z; d.w += s.w;
                px[j] = d;
                ub[2 * j]     = pack2(d.x, d.y);
                ub[2 * j + 1] = pack2(d.z, d.w);
                pa[j] = z4;
            }
#pragma unroll
            for (int j = 0; j < 4; ++j)
                pb[j] = make_uint4(ub[4 * j], ub[4 * j + 1], ub[4 * j + 2], ub[4 * j + 3]);
        }
    }
}

// ---------------------------------------------------------------------------
// Init: seed working state from inputs; zero agg.
// ---------------------------------------------------------------------------
__global__ void init_kernel(const float* __restrict__ x_in, const float* __restrict__ ea_in,
                            float* __restrict__ x_out, float* __restrict__ e_out,
                            unsigned short* __restrict__ x_bf, float* __restrict__ agg,
                            long long nTot, long long eTot)
{
    long long i = (long long)blockIdx.x * blockDim.x + threadIdx.x;
    const long long stride = (long long)gridDim.x * blockDim.x;
    for (; i < eTot; i += stride) {
        e_out[i] = ea_in[i];
        if (i < nTot) {
            const float v = x_in[i];
            x_out[i] = v;
            x_bf[i]  = f2bf(v);
            agg[i]   = 0.0f;
        }
    }
}

// Pack fp32 weights [L][K][64] -> fragment-major pair-packed bf16.
// dst frag (t,nt): 256 uints; lane L's 8 uints contiguous at [lane*8 + v].
__global__ void pack_frag_kernel(const float* __restrict__ src, unsigned* __restrict__ dst,
                                 int ktiles, int total)
{
    const int i = blockIdx.x * blockDim.x + threadIdx.x;
    if (i >= total) return;
    const int perLayer = ktiles * 4 * 256;
    const int l = i / perLayer;
    const int rem = i - l * perLayer;
    const int f = rem >> 8;
    const int lane = (rem >> 3) & 31;
    const int v = rem & 7;
    const int t = f >> 2, nt = f & 3;
    const int n15 = lane & 15, hh = lane >> 4;
    const int koff = (v < 4) ? (hh * 8 + 2 * v) : (16 + hh * 8 + 2 * (v - 4));
    const int k = t * 32 + koff;
    const int n = nt * 16 + n15;
    const long long base = (long long)l * (ktiles * 32) * 64;
    dst[i] = pack2(src[base + (long long)k * 64 + n],
                   src[base + (long long)(k + 1) * 64 + n]);
}

// ---------------------------------------------------------------------------
extern "C" void kernel_launch(void* const* d_in, const int* in_sizes, int n_in,
                              void* d_out, int out_size, void* d_ws, size_t ws_size,
                              hipStream_t stream)
{
    const float*     x    = (const float*)d_in[0];
    const long long* ei   = (const long long*)d_in[1];
    const float*     ea   = (const float*)d_in[2];
    const float*     e_w1 = (const float*)d_in[3];
    const float*     e_b1 = (const float*)d_in[4];
    const float*     e_w2 = (const float*)d_in[5];
    const float*     e_b2 = (const float*)d_in[6];
    const float*     e_g  = (const float*)d_in[7];
    const float*     e_be = (const float*)d_in[8];
    const float*     n_w1 = (const float*)d_in[9];
    const float*     n_b1 = (const float*)d_in[10];
    const float*     n_w2 = (const float*)d_in[11];
    const float*     n_b2 = (const float*)d_in[12];
    const float*     n_g  = (const float*)d_in[13];
    const float*     n_be = (const float*)d_in[14];

    const long long Nn  = (long long)in_sizes[0] / DF;   // 50000
    const long long Ee  = (long long)in_sizes[2] / DF;   // 800000
    const long long N64 = Nn * DF, E64 = Ee * DF;

    float* x_out = (float*)d_out;          // evolving node features [N,64]
    float* e_out = x_out + N64;            // evolving edge features [E,64]

    char* ws = (char*)d_ws;
    float* agg = (float*)ws;                       ws += ((N64 * 4 + 255) / 256) * 256;
    unsigned short* x_bf = (unsigned short*)ws;    ws += ((N64 * 2 + 255) / 256) * 256;
    unsigned* ew1p = (unsigned*)ws;                ws += LAYERS * 24 * 256 * 4;
    unsigned* ew2p = (unsigned*)ws;                ws += LAYERS * 8 * 256 * 4;
    unsigned* nw1p = (unsigned*)ws;                ws += LAYERS * 16 * 256 * 4;
    unsigned* nw2p = (unsigned*)ws;

    init_kernel<<<2048, 256, 0, stream>>>(x, ea, x_out, e_out, x_bf, agg, N64, E64);
    pack_frag_kernel<<<(LAYERS * 24 * 256 + 255) / 256, 256, 0, stream>>>(e_w1, ew1p, 6, LAYERS * 24 * 256);
    pack_frag_kernel<<<(LAYERS * 8  * 256 + 255) / 256, 256, 0, stream>>>(e_w2, ew2p, 2, LAYERS * 8 * 256);
    pack_frag_kernel<<<(LAYERS * 16 * 256 + 255) / 256, 256, 0, stream>>>(n_w1, nw1p, 4, LAYERS * 16 * 256);
    pack_frag_kernel<<<(LAYERS * 8  * 256 + 255) / 256, 256, 0, stream>>>(n_w2, nw2p, 2, LAYERS * 8 * 256);

    const int eblocks = (int)((Ee + 63) / 64);
    const int nblocks = (int)((Nn + 63) / 64);
    for (int l = 0; l < LAYERS; ++l) {
        edge_kernel<<<eblocks, 128, 0, stream>>>(
            ei, x_bf, e_out, agg,
            ew1p + (long long)l * 24 * 256, ew2p + (long long)l * 8 * 256,
            e_b1 + l * 64, e_b2 + l * 64, e_g + l * 64, e_be + l * 64, Ee);
        node_kernel<<<nblocks, 128, 0, stream>>>(
            x_out, x_bf, agg,
            nw1p + (long long)l * 16 * 256, nw2p + (long long)l * 8 * 256,
            n_b1 + l * 64, n_b2 + l * 64, n_g + l * 64, n_be + l * 64, Nn);
    }
}
